// LSTMNN_44057774522415
// MI455X (gfx1250) — matrix operand order repaired and checked by reading, not compiled, not run on hardware
//
#include <hip/hip_runtime.h>
#include <math.h>

// ---------------------------------------------------------------------------
// MI455X (gfx1250) implementation: BiLSTM + Child-Sum Tree-LSTM
// B=256, T=128, N=32768, E=300(->320), H=50, 4H=200, OUT=50
// GEMMs via v_wmma_f32_16x16x32_f16; tile staging via async global->LDS
// (ASYNCcnt) when the toolchain exposes the gfx1250 builtins.
// ---------------------------------------------------------------------------

typedef _Float16 f16;
typedef __attribute__((ext_vector_type(16))) _Float16 v16h;
typedef __attribute__((ext_vector_type(8)))  float    v8f;

union FragH { v16h v; uint4 q[2]; };

#if __has_builtin(__builtin_amdgcn_global_load_async_to_lds_b128) && \
    __has_builtin(__builtin_amdgcn_s_wait_asynccnt)
#define USE_ASYNC_LDS 1
#else
#define USE_ASYNC_LDS 0
#endif

#if USE_ASYNC_LDS
// Parameter type per hipcc diagnostic: pointer to int __vector_size__(16),
// global (AS1) source and LDS (AS3) destination.
typedef int vi4 __attribute__((vector_size(4 * sizeof(int))));
typedef __attribute__((address_space(1))) vi4 vi4_g;
typedef __attribute__((address_space(3))) vi4 vi4_l;
#endif

// 16-byte global -> LDS copy. Async (no VGPR data path) when available.
__device__ __forceinline__ void async_cp16(const f16* g, f16* l) {
#if USE_ASYNC_LDS
  __builtin_amdgcn_global_load_async_to_lds_b128((vi4_g*)g, (vi4_l*)l, 0, 0);
#else
  *(uint4*)l = *(const uint4*)g;
#endif
}

template <int N> __device__ __forceinline__ void async_wait() {
#if USE_ASYNC_LDS
  __builtin_amdgcn_s_wait_asynccnt(N);
#endif
}

__device__ __forceinline__ float sigm(float x) { return 1.f / (1.f + __expf(-x)); }

__device__ __forceinline__ v8f wmma16(const v16h& a, const v16h& b, v8f c) {
  // D = A(16x32 f16) * B(32x16 f16) + C(16x16 f32)
  return __builtin_amdgcn_wmma_f32_16x16x32_f16(false, a, false, b, (short)0, c, false, false);
}

// ---------------------------------------------------------------------------
// Weight prep: build f16 B-matrices (stored transposed: [n][Kpad], K contiguous)
// WT1 [512][320] = concat(Wih_f, Wih_b) ; B1[512] = concat(bias_f, bias_b)
// WT2 [256][128] = W_iou               ; B2[256] = b_iou (folded into iou0)
// UFT [64][64]   = U_f                 ; UIT [160][64] = U_iou
// ---------------------------------------------------------------------------
__global__ void k_prep(const float* Wih_f, const float* Wih_b,
                       const float* bias_f, const float* bias_b,
                       const float* W_iou, const float* b_iou,
                       const float* U_f, const float* U_iou,
                       f16* WT1, float* B1, f16* WT2, float* B2,
                       f16* UFT, f16* UIT) {
  int stride = gridDim.x * blockDim.x;
  int tid0 = blockIdx.x * blockDim.x + threadIdx.x;
  for (int i = tid0; i < 512 * 320; i += stride) {
    int n = i / 320, k = i % 320;
    float v = 0.f;
    if (k < 300) {
      if (n < 200) v = Wih_f[n * 300 + k];
      else if (n < 400) v = Wih_b[(n - 200) * 300 + k];
    }
    WT1[i] = (f16)v;
  }
  for (int i = tid0; i < 512; i += stride)
    B1[i] = (i < 200) ? bias_f[i] : (i < 400) ? bias_b[i - 200] : 0.f;
  for (int i = tid0; i < 256 * 128; i += stride) {
    int n = i / 128, k = i % 128;
    WT2[i] = (f16)((n < 150 && k < 100) ? W_iou[n * 100 + k] : 0.f);
  }
  for (int i = tid0; i < 256; i += stride) B2[i] = (i < 150) ? b_iou[i] : 0.f;
  for (int i = tid0; i < 64 * 64; i += stride) {
    int n = i / 64, k = i % 64;
    UFT[i] = (f16)((n < 50 && k < 50) ? U_f[n * 50 + k] : 0.f);
  }
  for (int i = tid0; i < 160 * 64; i += stride) {
    int n = i / 64, k = i % 64;
    UIT[i] = (f16)((n < 150 && k < 50) ? U_iou[n * 50 + k] : 0.f);
  }
}

// ---------------------------------------------------------------------------
// Embedding gather -> XH [32768][320] f16 (zero-padded); zero RH16 and H16
// ---------------------------------------------------------------------------
__global__ void k_gather(const int* ids, const float* emb, f16* XH, f16* RH, f16* H16) {
  int stride = gridDim.x * blockDim.x;
  int tid0 = blockIdx.x * blockDim.x + threadIdx.x;
  for (int i = tid0; i < 32768 * 320; i += stride) {
    int row = i / 320, k = i % 320;
    float v = (k < 300) ? emb[ids[row] * 300 + k] : 0.f;
    XH[i] = (f16)v;
  }
  for (int i = tid0; i < 32768 * 128; i += stride) RH[i] = (f16)0.f;
  for (int i = tid0; i < 32768 * 64; i += stride) H16[i] = (f16)0.f;
}

// ---------------------------------------------------------------------------
// Generic GEMM: C[M][ldc] = A[M][Kp]f16 * BT[n][Kp]f16 (+bias), f32 out.
// Workgroup tile 64(M) x 128(N); 8 waves; each wave owns 4 16x16 C tiles.
// Double-buffered LDS; tiles streamed with GLOBAL_LOAD_ASYNC_TO_LDS_B128.
// ---------------------------------------------------------------------------
__global__ void __launch_bounds__(256) k_gemm(const f16* __restrict__ A,
                                              const f16* __restrict__ BT,
                                              const float* __restrict__ bias,
                                              float* __restrict__ C,
                                              int Kp, int Nact, int ldc) {
  __shared__ alignas(16) f16 At[2][64 * 32];
  __shared__ alignas(16) f16 Bt[2][128 * 32];
  int tid = threadIdx.x;
  int lane = tid & 31, w = tid >> 5;
  int mt = w & 3, ng = w >> 2;            // wave -> (M-tile, N-cluster of 4 tiles)
  int la = lane & 15, lh = lane >> 4;
  long long Mblk = (long long)blockIdx.x * 64;
  int Nblk = blockIdx.y * 128;
  v8f acc[4] = {};

  // Per k-chunk stage = 3 async b128 ops per thread (1 for A, 2 for B).
  auto stage = [&](int buf, int kk) {
    int r = tid >> 2, q = tid & 3;
    async_cp16(A + (Mblk + r) * Kp + kk + q * 8, At[buf] + r * 32 + q * 8);
    for (int rr = 0; rr < 2; ++rr) {
      int idx = tid + rr * 256;
      int r2 = idx >> 2, q2 = idx & 3;
      async_cp16(BT + (long long)(Nblk + r2) * Kp + kk + q2 * 8,
                 Bt[buf] + r2 * 32 + q2 * 8);
    }
  };

  int nk = Kp >> 5;
  stage(0, 0);
  for (int ki = 0; ki < nk; ++ki) {
    int cur = ki & 1;
    if (ki + 1 < nk) {
      stage(cur ^ 1, (ki + 1) * 32);
      async_wait<3>();   // in-order: the 3 older ops (current buffer) retired
    } else {
      async_wait<0>();
    }
    __syncthreads();
    FragH a; // A layout: lane row=la, elems 0-7 -> K=lh*8.., elems 8-15 -> K=16+lh*8..
    const f16* ap = At[cur] + (mt * 16 + la) * 32 + lh * 8;
    a.q[0] = *(const uint4*)ap;
    a.q[1] = *(const uint4*)(ap + 16);
    for (int bt = 0; bt < 4; ++bt) { // B layout: lane col=la, elems 0-7 -> K=lh*8.., elems 8-15 -> K=16+lh*8..
      FragH b;
      const f16* bp = Bt[cur] + ((ng * 4 + bt) * 16 + la) * 32 + lh * 8;
      b.q[0] = *(const uint4*)bp;
      b.q[1] = *(const uint4*)(bp + 16);
      acc[bt] = wmma16(a.v, b.v, acc[bt]);
    }
    __syncthreads();
  }
  for (int bt = 0; bt < 4; ++bt) {
    int col = Nblk + (ng * 4 + bt) * 16 + la;
    if (col < Nact) {
      float bv = bias[col];
      for (int r = 0; r < 8; ++r) {
        long long row = Mblk + mt * 16 + r + lh * 8; // C: M = r + 8*(lane>=16)
        C[row * ldc + col] = acc[bt][r] + bv;
      }
    }
  }
}

// ---------------------------------------------------------------------------
// Sequential BiLSTM. One block per (direction, 16-batch-row tile) = 32 blocks.
// 13 waves: wave w owns gate N-tile w (cols w*16..w*16+15 of 200).
// Whh^T resident in LDS; h/c state in LDS; 128 steps in-kernel.
// ---------------------------------------------------------------------------
__global__ void __launch_bounds__(416) k_lstm(const float* __restrict__ Whh_f,
                                              const float* __restrict__ Whh_b,
                                              const float* __restrict__ XW,
                                              f16* __restrict__ RH) {
  __shared__ alignas(16) f16 WT[208 * 64];   // B-layout [n][64], n>=200 / k>=50 zero
  __shared__ alignas(16) f16 hA[16 * 64];    // A tile of h (cols 50..63 zero)
  __shared__ float gb[16 * 208];             // gate pre-activations
  __shared__ float cst[16 * 50];             // cell state
  int tid = threadIdx.x;
  int d = blockIdx.x >> 4, b0 = (blockIdx.x & 15) * 16;
  const float* Whh = d ? Whh_b : Whh_f;
  for (int i = tid; i < 208 * 64; i += 416) {
    int n = i >> 6, k = i & 63;
    WT[i] = (f16)((n < 200 && k < 50) ? Whh[n * 50 + k] : 0.f);
  }
  for (int i = tid; i < 16 * 64; i += 416) hA[i] = (f16)0.f;
  for (int i = tid; i < 16 * 50; i += 416) cst[i] = 0.f;
  __syncthreads();

  int lane = tid & 31, w = tid >> 5;  // w = N-tile 0..12
  int la = lane & 15, lh = lane >> 4;
  for (int s = 0; s < 128; ++s) {
    int t = d ? (127 - s) : s;
    v8f acc = {};
    for (int kc = 0; kc < 64; kc += 32) {
      FragH a, b;
      const f16* ap = hA + la * 64 + kc + lh * 8;
      a.q[0] = *(const uint4*)ap;
      a.q[1] = *(const uint4*)(ap + 16);
      const f16* bp = WT + (w * 16 + la) * 64 + kc + lh * 8;
      b.q[0] = *(const uint4*)bp;
      b.q[1] = *(const uint4*)(bp + 16);
      acc = wmma16(a.v, b.v, acc);
    }
    int col = w * 16 + la;
    if (col < 200) {
      for (int r = 0; r < 8; ++r) {
        int row = r + lh * 8;
        long long tok = (long long)(b0 + row) * 128 + t;
        gb[row * 208 + col] = acc[r] + XW[tok * 400 + d * 200 + col]; // bias folded in XW
      }
    }
    __syncthreads();
    for (int it = tid; it < 800; it += 416) { // gates: i,f,g,o (PyTorch order)
      int i = it / 50, j = it % 50;
      float gi = gb[i * 208 + j];
      float gf = gb[i * 208 + 50 + j];
      float gg = gb[i * 208 + 100 + j];
      float go = gb[i * 208 + 150 + j];
      float c = sigm(gf) * cst[i * 50 + j] + sigm(gi) * tanhf(gg);
      float h = sigm(go) * tanhf(c);
      cst[i * 50 + j] = c;
      hA[i * 64 + j] = (f16)h;
      long long tok = (long long)(b0 + i) * 128 + t;
      RH[tok * 128 + d * 50 + j] = (f16)h; // fwd cols 0..49, bwd 50..99
    }
    __syncthreads();
  }
}

// ---------------------------------------------------------------------------
// Tree level kernel. Block handles 16 nodes of level L (complete binary tree:
// node p has children 2p+1, 2p+2 if < 128). Children were finalized at L+1.
//   fg   = children_h @ U_f^T           (2 M-tiles x 4 N-tiles)
//   ioub = (h_c1+h_c2) @ U_iou^T        (1 M-tile x 10 N-tiles)
// Epilogue: gates, update c / h / h16.
// ---------------------------------------------------------------------------
__global__ void __launch_bounds__(256) k_tree(int L, int nL,
                                              f16* __restrict__ H16,
                                              const float* __restrict__ Y,
                                              const f16* __restrict__ UFT,
                                              const f16* __restrict__ UIT,
                                              const float* __restrict__ b_f,
                                              float* __restrict__ Cst,
                                              float* __restrict__ Hout) {
  __shared__ alignas(16) f16 hsA[16 * 64];
  __shared__ alignas(16) f16 chA[32 * 64];
  __shared__ alignas(16) f16 Uf[64 * 64];
  __shared__ alignas(16) f16 Ui[160 * 64];
  __shared__ float fg[32 * 64];
  __shared__ float ioub[16 * 160];
  int tid = threadIdx.x;
  int r0 = blockIdx.x * 16;
  int M = 256 * nL;
  int pbase = (1 << L) - 1;

  // Weight staging: straight copies -> async global->LDS path.
  for (int i = tid; i < (64 * 64) / 8; i += 256)
    async_cp16(UFT + i * 8, Uf + i * 8);
  for (int i = tid; i < (160 * 64) / 8; i += 256)
    async_cp16(UIT + i * 8, Ui + i * 8);

  for (int idx = tid; idx < 16 * 64; idx += 256) { // h_sum A-tile
    int i = idx >> 6, k = idx & 63;
    int r = r0 + i;
    float v = 0.f;
    if (r < M) {
      int b = r / nL, p = pbase + (r % nL);
      int c1 = 2 * p + 1, c2 = 2 * p + 2;
      if (c1 < 128) v += (float)H16[(b * 128 + c1) * 64 + k];
      if (c2 < 128) v += (float)H16[(b * 128 + c2) * 64 + k];
    }
    hsA[idx] = (f16)v;
  }
  for (int idx = tid; idx < 32 * 64; idx += 256) { // children A-tiles
    int row = idx >> 6, k = idx & 63;
    int i = row >> 1, which = row & 1;
    int r = r0 + i;
    f16 v = (f16)0.f;
    if (r < M) {
      int b = r / nL, p = pbase + (r % nL);
      int c = 2 * p + 1 + which;
      if (c < 128) v = H16[(b * 128 + c) * 64 + k];
    }
    chA[idx] = v;
  }
  async_wait<0>();
  __syncthreads();

  int lane = tid & 31, w = tid >> 5;
  int la = lane & 15, lh = lane >> 4;
  for (int f = w; f < 18; f += 8) { // 18 output fragments over 8 waves
    v8f acc = {};
    if (f < 8) {
      int mt = f & 1, nt = f >> 1;
      for (int kc = 0; kc < 64; kc += 32) {
        FragH a, b;
        const f16* ap = chA + (mt * 16 + la) * 64 + kc + lh * 8;
        a.q[0] = *(const uint4*)ap; a.q[1] = *(const uint4*)(ap + 16);
        const f16* bp = Uf + (nt * 16 + la) * 64 + kc + lh * 8;
        b.q[0] = *(const uint4*)bp; b.q[1] = *(const uint4*)(bp + 16);
        acc = wmma16(a.v, b.v, acc);
      }
      for (int r = 0; r < 8; ++r)
        fg[(mt * 16 + r + lh * 8) * 64 + nt * 16 + la] = acc[r];
    } else {
      int nt = f - 8;
      for (int kc = 0; kc < 64; kc += 32) {
        FragH a, b;
        const f16* ap = hsA + la * 64 + kc + lh * 8;
        a.q[0] = *(const uint4*)ap; a.q[1] = *(const uint4*)(ap + 16);
        const f16* bp = Ui + (nt * 16 + la) * 64 + kc + lh * 8;
        b.q[0] = *(const uint4*)bp; b.q[1] = *(const uint4*)(bp + 16);
        acc = wmma16(a.v, b.v, acc);
      }
      for (int r = 0; r < 8; ++r)
        ioub[(r + lh * 8) * 160 + nt * 16 + la] = acc[r];
    }
  }
  __syncthreads();

  for (int it = tid; it < 800; it += 256) {
    int i = it / 50, j = it % 50;
    int r = r0 + i;
    if (r >= M) continue;
    int b = r / nL, p = pbase + (r % nL);
    int g = b * 128 + p;
    int c1 = 2 * p + 1, c2 = 2 * p + 2;
    float bf = b_f[j];
    float csum = 0.f;
    if (c1 < 128) csum += sigm(fg[(2 * i) * 64 + j] + bf) * Cst[(b * 128 + c1) * 50 + j];
    if (c2 < 128) csum += sigm(fg[(2 * i + 1) * 64 + j] + bf) * Cst[(b * 128 + c2) * 50 + j];
    float ii = ioub[i * 160 + j]        + Y[g * 160 + j];        // Y has b_iou folded
    float oo = ioub[i * 160 + 50 + j]   + Y[g * 160 + 50 + j];
    float uu = ioub[i * 160 + 100 + j]  + Y[g * 160 + 100 + j];
    float cn = sigm(ii) * tanhf(uu) + csum;
    float hn = sigm(oo) * tanhf(cn);
    Cst[g * 50 + j] = cn;
    Hout[g * 50 + j] = hn;
    H16[g * 64 + j] = (f16)hn;
  }
}

// ---------------------------------------------------------------------------
extern "C" void kernel_launch(void* const* d_in, const int* in_sizes, int n_in,
                              void* d_out, int out_size, void* d_ws, size_t ws_size,
                              hipStream_t stream) {
  const int*   ids    = (const int*)  d_in[0];
  const float* emb    = (const float*)d_in[5];
  const float* Wih_f  = (const float*)d_in[6];
  const float* Whh_f  = (const float*)d_in[7];
  const float* bias_f = (const float*)d_in[8];
  const float* Wih_b  = (const float*)d_in[9];
  const float* Whh_b  = (const float*)d_in[10];
  const float* bias_b = (const float*)d_in[11];
  const float* W_iou  = (const float*)d_in[12];
  const float* U_iou  = (const float*)d_in[13];
  const float* b_iou  = (const float*)d_in[14];
  const float* U_f    = (const float*)d_in[15];
  const float* b_f    = (const float*)d_in[16];
  // d_in[17]=W_hid, d_in[18]=b_hid: dead code (every node's h is overwritten
  // at its own tree level before ever being read).

  char* ws = (char*)d_ws;
  f16*   XH  = (f16*)  (ws + 0);          // [32768][320] f16
  float* XW  = (float*)(ws + 20971520);   // [32768][400] f32 (x@Wih^T + bias, both dirs)
  f16*   RH  = (f16*)  (ws + 73400320);   // [32768][128] f16 rnn_out (K-padded)
  float* Y   = (float*)(ws + 81788928);   // [32768][160] f32 iou0 + b_iou
  f16*   H16 = (f16*)  (ws + 102760448);  // [32768][64]  f16 node h
  float* CC  = (float*)(ws + 106954752);  // [32768][50]  f32 node c
  f16*   WT1 = (f16*)  (ws + 113508352);  // [512][320]
  float* B1  = (float*)(ws + 113836032);  // [512]
  f16*   WT2 = (f16*)  (ws + 113838080);  // [256][128]
  float* B2  = (float*)(ws + 113903616);  // [256]
  f16*   UFT = (f16*)  (ws + 113904640);  // [64][64]
  f16*   UIT = (f16*)  (ws + 113912832);  // [160][64]

  k_prep<<<64, 256, 0, stream>>>(Wih_f, Wih_b, bias_f, bias_b, W_iou, b_iou,
                                 U_f, U_iou, WT1, B1, WT2, B2, UFT, UIT);
  k_gather<<<2048, 256, 0, stream>>>(ids, emb, XH, RH, H16);

  dim3 g1(512, 4);  // [32768x320] x [320x400]
  k_gemm<<<g1, 256, 0, stream>>>(XH, WT1, B1, XW, 320, 400, 400);

  k_lstm<<<32, 416, 0, stream>>>(Whh_f, Whh_b, XW, RH);

  dim3 g2(512, 2);  // [32768x128] x [128x150]
  k_gemm<<<g2, 256, 0, stream>>>(RH, WT2, B2, Y, 128, 150, 160);

  for (int L = 7; L >= 0; --L) {
    int nL = (L < 7) ? (1 << L) : 1;       // depth-7 level is the single node 127
    int blocks = (256 * nL + 15) / 16;
    k_tree<<<blocks, 256, 0, stream>>>(L, nL, H16, Y, UFT, UIT, b_f, CC, (float*)d_out);
  }
}
